// ContextEncodingTransformer_26594437496911
// MI455X (gfx1250) — compile-verified, matrix operand order checked by reading, not run on hardware
//
#include <hip/hip_runtime.h>

typedef __attribute__((ext_vector_type(16))) __bf16 v16bf;
typedef __attribute__((ext_vector_type(8)))  __bf16 v8bf;
typedef __attribute__((ext_vector_type(4)))  __bf16 v4bf;
typedef __attribute__((ext_vector_type(8)))  float  v8f;
typedef __attribute__((ext_vector_type(4), aligned(4))) float v4f_u;
typedef __attribute__((ext_vector_type(4)))  unsigned int v4u;
typedef __attribute__((ext_vector_type(4)))  int v4i;
typedef __attribute__((address_space(1))) v4i gv4i;   // global int4
typedef __attribute__((address_space(3))) v4i lv4i;   // LDS int4

#define NFC  256
#define NN   12
#define BT   64
#define SS   3600          // 45*80
#define ROWS (BT*NN)       // 768

#if __has_builtin(__builtin_amdgcn_global_load_async_to_lds_b128)
#define HAVE_ASYNC_LDS 1
#else
#define HAVE_ASYNC_LDS 0
#endif

__device__ __forceinline__ v16bf cat8(v8bf lo, v8bf hi) {
  return __builtin_shufflevector(lo, hi, 0,1,2,3,4,5,6,7,8,9,10,11,12,13,14,15);
}

// ---------------------------------------------------------------------------
// One-time weight split: W f32 [N][K] -> bf16 hi plane + bf16 lo plane.
// ---------------------------------------------------------------------------
__global__ __launch_bounds__(256)
void split_bf16_kernel(const float* __restrict__ W, __bf16* __restrict__ H,
                       __bf16* __restrict__ L, int n)
{
  const int i = (blockIdx.x * 256 + threadIdx.x) * 4;
  if (i + 3 < n) {
    const v4f_u v = *(const v4f_u*)(W + i);
    v4bf h4, l4;
    #pragma unroll
    for (int j = 0; j < 4; ++j) {
      const __bf16 hh = (__bf16)v[j];
      h4[j] = hh;
      l4[j] = (__bf16)(v[j] - (float)hh);
    }
    *(v4bf*)(H + i) = h4;
    *(v4bf*)(L + i) = l4;
  }
}

// ---------------------------------------------------------------------------
// C[M x 256] = A[M x K] * B[256 x K]^T + bias, bf16x3 split WMMA
// (hi*hi + hi*lo + lo*hi). B supplied as pre-split bf16 hi/lo planes [N][K].
// AMODE: 0 = A row-major [M][K]
//        1 = image_feature: m=(b,s), addr = b*K*SS + k*SS + s
//        2 = roi_feature:   m=(rr,p), addr = rr*K*25 + k*25 + p
// OMODE: 0 = C row-major [M][256];  2 = addr = rr*6400 + n*25 + p
// Block: 256 threads (8 waves), tile 32(M) x 256(N), K-step 32,
// double-buffered LDS, async global->LDS B-tile fill.
// ---------------------------------------------------------------------------
template <int AMODE, int OMODE>
__global__ __launch_bounds__(256)
void gemm_wmma_bf16x3(const float* __restrict__ A,
                      const __bf16* __restrict__ Bhg, const __bf16* __restrict__ Blg,
                      const float* __restrict__ bias, float* __restrict__ C,
                      int M, int K)
{
  __shared__ __align__(16) __bf16 Ah[2][32][40];
  __shared__ __align__(16) __bf16 Al[2][32][40];
  __shared__ __align__(16) __bf16 Bh[2][256][40];
  __shared__ __align__(16) __bf16 Bl[2][256][40];

  const int tid   = threadIdx.x;
  const int lane  = tid & 31;
  const int wave  = tid >> 5;
  const int wm    = wave >> 2;        // 0..1 : 16-row slab
  const int wn    = wave & 3;         // 0..3 : 64-col slab
  const int m0    = blockIdx.x * 32;
  const int row16 = lane & 15;
  const int kb    = (lane >> 4) * 8;  // lanes 0-15: K 0..7/16..23 ; 16-31: K 8..15/24..31

  v8f acc[4];
  #pragma unroll
  for (int s = 0; s < 4; ++s)
    #pragma unroll
    for (int j = 0; j < 8; ++j) acc[s][j] = 0.0f;

  // ---- per-thread A quad addressing, hoisted out of the K loop ----
  int ar_, ac_;
  long aoff1, aoff2, aoff3, aadv;
  const float* abase;
  if (AMODE == 0) {
    ar_ = tid >> 3; ac_ = (tid & 7) * 4;           // 4 consecutive k
    abase = A + (long)(m0 + ar_) * K + ac_;
    aoff1 = 1; aoff2 = 2; aoff3 = 3;
    aadv = 32;
  } else {
    ar_ = (tid & 7) * 4; ac_ = tid >> 3;           // 4 consecutive m
    long off[4];
    #pragma unroll
    for (int j = 0; j < 4; ++j) {
      const int m = m0 + ar_ + j;
      if (AMODE == 1) { const int b = m / SS, s = m - b * SS; off[j] = (long)b * K * SS + (long)ac_ * SS + s; }
      else            { const int rr = m / 25, p = m - rr * 25; off[j] = (long)rr * K * 25 + (long)ac_ * 25 + p; }
    }
    abase = A + off[0];
    aoff1 = off[1] - off[0]; aoff2 = off[2] - off[0]; aoff3 = off[3] - off[0];
    aadv  = (AMODE == 1) ? 32L * SS : 32L * 25;
  }
  const bool afast = (aoff1 == 1) && (aoff2 == 2) && (aoff3 == 3);

  // ---- per-thread B segment map: 2 planes x 256 rows x 4 x 16B = 8 segs ----
  int bn_[8], bsg_[8], bpl_[8];
  #pragma unroll
  for (int j = 0; j < 8; ++j) {
    const int gid = j * 256 + tid;
    bpl_[j] = gid >> 10;
    const int rem = gid & 1023;
    bn_[j] = rem >> 2; bsg_[j] = rem & 3;
  }

  float astage[4];
  v4u   bstage[8];
  (void)bstage;

  auto issue_tile = [&](int t1, int bufn) {
    const long kofs = (long)t1 * 32;
    #pragma unroll
    for (int j = 0; j < 8; ++j) {
      const __bf16* gp = (bpl_[j] ? Blg : Bhg) + (long)bn_[j] * K + kofs + bsg_[j] * 8;
#if HAVE_ASYNC_LDS
      __bf16* lp = bpl_[j] ? &Bl[bufn][bn_[j]][bsg_[j] * 8] : &Bh[bufn][bn_[j]][bsg_[j] * 8];
      __builtin_amdgcn_global_load_async_to_lds_b128((gv4i*)(void*)gp, (lv4i*)lp, 0, 0);
#else
      bstage[j] = *(const v4u*)gp;
#endif
    }
    const float* ap = abase + (long)t1 * aadv;
    if (afast) {
      const v4f_u v = *(const v4f_u*)ap;
      astage[0] = v[0]; astage[1] = v[1]; astage[2] = v[2]; astage[3] = v[3];
    } else {
      astage[0] = ap[0]; astage[1] = ap[aoff1]; astage[2] = ap[aoff2]; astage[3] = ap[aoff3];
    }
  };

  auto store_tile = [&](int bufn) {
#if !HAVE_ASYNC_LDS
    #pragma unroll
    for (int j = 0; j < 8; ++j) {
      __bf16* lp = bpl_[j] ? &Bl[bufn][bn_[j]][bsg_[j] * 8] : &Bh[bufn][bn_[j]][bsg_[j] * 8];
      *(v4u*)lp = bstage[j];
    }
#endif
    if (AMODE == 0) {                 // 4 consecutive k -> vector LDS store
      v4bf h4, l4;
      #pragma unroll
      for (int j = 0; j < 4; ++j) {
        const __bf16 hh = (__bf16)astage[j];
        h4[j] = hh; l4[j] = (__bf16)(astage[j] - (float)hh);
      }
      *(v4bf*)&Ah[bufn][ar_][ac_] = h4;
      *(v4bf*)&Al[bufn][ar_][ac_] = l4;
    } else {                          // 4 consecutive m rows, same k
      #pragma unroll
      for (int j = 0; j < 4; ++j) {
        const __bf16 hh = (__bf16)astage[j];
        Ah[bufn][ar_ + j][ac_] = hh;
        Al[bufn][ar_ + j][ac_] = (__bf16)(astage[j] - (float)hh);
      }
    }
  };

  auto compute_tile = [&](int bufn) {
    const int arr = wm * 16 + row16;
    const v16bf a_hi = cat8(*(const v8bf*)&Ah[bufn][arr][kb], *(const v8bf*)&Ah[bufn][arr][16 + kb]);
    const v16bf a_lo = cat8(*(const v8bf*)&Al[bufn][arr][kb], *(const v8bf*)&Al[bufn][arr][16 + kb]);
    #pragma unroll
    for (int sub = 0; sub < 4; ++sub) {
      const int bnn = wn * 64 + sub * 16 + row16;
      const v16bf b_hi = cat8(*(const v8bf*)&Bh[bufn][bnn][kb], *(const v8bf*)&Bh[bufn][bnn][16 + kb]);
      const v16bf b_lo = cat8(*(const v8bf*)&Bl[bufn][bnn][kb], *(const v8bf*)&Bl[bufn][bnn][16 + kb]);
      acc[sub] = __builtin_amdgcn_wmma_f32_16x16x32_bf16(false, a_hi, false, b_hi, (short)0, acc[sub], false, false);
      acc[sub] = __builtin_amdgcn_wmma_f32_16x16x32_bf16(false, a_hi, false, b_lo, (short)0, acc[sub], false, false);
      acc[sub] = __builtin_amdgcn_wmma_f32_16x16x32_bf16(false, a_lo, false, b_hi, (short)0, acc[sub], false, false);
    }
  };

  // ---- software-pipelined main loop, one barrier per K-step ----
  const int T = K >> 5;
  issue_tile(0, 0);
  store_tile(0);
  for (int t = 0; t < T; ++t) {
    const int buf = t & 1;
#if HAVE_ASYNC_LDS
    asm volatile("s_wait_asynccnt 0x0" ::: "memory");
#endif
    __syncthreads();
    if (t + 1 < T) issue_tile(t + 1, buf ^ 1);
    compute_tile(buf);
    if (t + 1 < T) store_tile(buf ^ 1);
  }

  // ---- epilogue: D layout lane = N%16, vgpr r -> M = r + 8*(lane/16) ----
  #pragma unroll
  for (int sub = 0; sub < 4; ++sub) {
    const int n  = wn * 64 + sub * 16 + row16;
    const float bv = bias[n];
    #pragma unroll
    for (int r = 0; r < 8; ++r) {
      const int m = m0 + wm * 16 + r + ((lane >> 4) << 3);
      long idx;
      if (OMODE == 0) idx = (long)m * 256 + n;
      else { const int rr = m / 25, p = m - rr * 25; idx = (long)rr * 6400 + (long)n * 25 + p; }
      C[idx] = acc[sub][r] + bv;
    }
  }
}

// ---------------------------------------------------------------------------
// Attention scores: a[b,n,s] = dot(emb[b,n,:], img[b,s,:])  (img stored [b][s][c])
// ---------------------------------------------------------------------------
__global__ __launch_bounds__(384)
void scores_kernel(const float* __restrict__ img, const float* __restrict__ emb,
                   float* __restrict__ sc)
{
  const int chunk = blockIdx.x;
  const int b     = blockIdx.y;
  const int tid   = threadIdx.x;
  __shared__ float se[NN][NFC];
  __shared__ float si[30][NFC + 1];

  for (int i = tid; i < NN * NFC; i += 384)
    se[i >> 8][i & 255] = emb[(long)(b * NN + (i >> 8)) * NFC + (i & 255)];

  for (int t = 0; t < 15; ++t) {
    const int s0 = chunk * 450 + t * 30;
    __syncthreads();
    for (int i = tid; i < 30 * NFC; i += 384)
      si[i >> 8][i & 255] = img[((long)b * SS + s0 + (i >> 8)) * NFC + (i & 255)];
    __syncthreads();
    if (tid < 360) {
      const int s = tid / 12, n = tid - 12 * (tid / 12);
      float acc = 0.0f;
      #pragma unroll 8
      for (int c = 0; c < NFC; ++c) acc += si[s][c] * se[n][c];
      sc[(long)(b * NN + n) * SS + s0 + s] = acc;
    }
  }
}

// ---------------------------------------------------------------------------
// In-place row softmax over 3600 elements. grid = 768 rows, 256 threads.
// ---------------------------------------------------------------------------
__global__ __launch_bounds__(256)
void softmax_kernel(float* __restrict__ sc)
{
  __shared__ float red[256];
  const int tid = threadIdx.x;
  float* row = sc + (long)blockIdx.x * SS;

  float m = -3.4e38f;
  for (int i = tid; i < SS; i += 256) m = fmaxf(m, row[i]);
  red[tid] = m; __syncthreads();
  for (int s = 128; s > 0; s >>= 1) { if (tid < s) red[tid] = fmaxf(red[tid], red[tid + s]); __syncthreads(); }
  m = red[0]; __syncthreads();

  float sum = 0.0f;
  for (int i = tid; i < SS; i += 256) { const float e = __expf(row[i] - m); row[i] = e; sum += e; }
  red[tid] = sum; __syncthreads();
  for (int s = 128; s > 0; s >>= 1) { if (tid < s) red[tid] += red[tid + s]; __syncthreads(); }
  const float inv = 1.0f / red[0];
  __syncthreads();
  for (int i = tid; i < SS; i += 256) row[i] *= inv;
}

// ---------------------------------------------------------------------------
// ctx[b,n,c] = sum_s A[b,n,s] * img[b,s,c]
// ---------------------------------------------------------------------------
__global__ __launch_bounds__(256)
void ctx_kernel(const float* __restrict__ sc, const float* __restrict__ img,
                float* __restrict__ ctx)
{
  const int cchunk = blockIdx.x;
  const int b   = blockIdx.y;
  const int tid = threadIdx.x;
  const int cl  = tid & 63;
  const int ng  = tid >> 6;          // n = ng*3 .. ng*3+2
  const int c0  = cchunk * 64;
  __shared__ float At[NN][121];
  float a0 = 0.0f, a1 = 0.0f, a2 = 0.0f;

  for (int st = 0; st < 30; ++st) {
    __syncthreads();
    for (int i = tid; i < NN * 120; i += 256) {
      const int n = i / 120, s = i - n * 120;
      At[n][s] = sc[(long)(b * NN + n) * SS + st * 120 + s];
    }
    __syncthreads();
    #pragma unroll 4
    for (int s = 0; s < 120; ++s) {
      const float v = img[((long)b * SS + st * 120 + s) * NFC + c0 + cl];
      a0 += At[ng * 3 + 0][s] * v;
      a1 += At[ng * 3 + 1][s] * v;
      a2 += At[ng * 3 + 2][s] * v;
    }
  }
  ctx[(long)(b * NN + ng * 3 + 0) * NFC + c0 + cl] = a0;
  ctx[(long)(b * NN + ng * 3 + 1) * NFC + c0 + cl] = a1;
  ctx[(long)(b * NN + ng * 3 + 2) * NFC + c0 + cl] = a2;
}

// ---------------------------------------------------------------------------
// Fused: x = LN(ctx+emb); y = relu(x W1^T + b1) W2^T + b2; out = LN(x+y)
// ---------------------------------------------------------------------------
__device__ __forceinline__ float block_sum(float v, float* red) {
  const int tid = threadIdx.x;
  red[tid] = v; __syncthreads();
  for (int s = 128; s > 0; s >>= 1) { if (tid < s) red[tid] += red[tid + s]; __syncthreads(); }
  const float r = red[0]; __syncthreads();
  return r;
}

__global__ __launch_bounds__(256)
void ffn_kernel(const float* __restrict__ ctx, const float* __restrict__ emb,
                const float* __restrict__ g1, const float* __restrict__ be1,
                const float* __restrict__ W1, const float* __restrict__ b1,
                const float* __restrict__ W2, const float* __restrict__ b2,
                const float* __restrict__ g2, const float* __restrict__ be2,
                float* __restrict__ out)
{
  __shared__ float xs[256];
  __shared__ float hs[256];
  __shared__ float red[256];
  const int r = blockIdx.x, c = threadIdx.x;

  const float t  = ctx[(long)r * 256 + c] + emb[(long)r * 256 + c];
  const float mu = block_sum(t, red) * (1.0f / 256.0f);
  const float d  = t - mu;
  const float var = block_sum(d * d, red) * (1.0f / 256.0f);
  const float x  = d * rsqrtf(var + 1e-5f) * g1[c] + be1[c];
  xs[c] = x; __syncthreads();

  float h = b1[c];
  #pragma unroll 8
  for (int j = 0; j < 256; ++j) h += xs[j] * W1[(long)c * 256 + j];
  h = fmaxf(h, 0.0f);
  hs[c] = h; __syncthreads();

  float y = b2[c];
  #pragma unroll 8
  for (int j = 0; j < 256; ++j) y += hs[j] * W2[(long)c * 256 + j];

  const float t2  = x + y;
  const float mu2 = block_sum(t2, red) * (1.0f / 256.0f);
  const float d2  = t2 - mu2;
  const float v2  = block_sum(d2 * d2, red) * (1.0f / 256.0f);
  out[(long)r * 256 + c] = d2 * rsqrtf(v2 + 1e-5f) * g2[c] + be2[c];
}

// ---------------------------------------------------------------------------
extern "C" void kernel_launch(void* const* d_in, const int* in_sizes, int n_in,
                              void* d_out, int out_size, void* d_ws, size_t ws_size,
                              hipStream_t stream) {
  const float* roi   = (const float*)d_in[0];   // [768,1024,5,5]
  const float* image = (const float*)d_in[1];   // [64,768,45,80]
  const float* w_ds1 = (const float*)d_in[2];   // [256,1024]
  const float* b_ds1 = (const float*)d_in[3];
  const float* w_ds2 = (const float*)d_in[4];   // [256,768]
  const float* b_ds2 = (const float*)d_in[5];
  const float* w_emb = (const float*)d_in[6];   // [256,6400]
  const float* b_emb = (const float*)d_in[7];
  const float* g1    = (const float*)d_in[8];
  const float* be1   = (const float*)d_in[9];
  const float* W1    = (const float*)d_in[10];
  const float* b1    = (const float*)d_in[11];
  const float* W2    = (const float*)d_in[12];
  const float* b2    = (const float*)d_in[13];
  const float* g2    = (const float*)d_in[14];
  const float* be2   = (const float*)d_in[15];
  float* out = (float*)d_out;

  // workspace layout (f32 scratch, then bf16 weight planes)
  float* ws    = (float*)d_ws;
  float* img   = ws;                                // [64*3600, 256]
  float* roids = img   + (long)BT * SS * NFC;       // [768, 6400]
  float* embw  = roids + (long)ROWS * NFC * 25;     // [768, 256]
  float* sc    = embw  + (long)ROWS * NFC;          // [768, 3600]
  float* ctxw  = sc    + (long)ROWS * SS;           // [768, 256]
  __bf16* ds2h = (__bf16*)(ctxw + (long)ROWS * NFC);
  __bf16* ds2l = ds2h + 256 * 768;
  __bf16* ds1h = ds2l + 256 * 768;
  __bf16* ds1l = ds1h + 256 * 1024;
  __bf16* embh = ds1l + 256 * 1024;
  __bf16* embl = embh + 256 * 6400;

  // 0) split weights into bf16 hi/lo planes (once, tiny)
  split_bf16_kernel<<<dim3(256 * 768  / 1024), 256, 0, stream>>>(w_ds2, ds2h, ds2l, 256 * 768);
  split_bf16_kernel<<<dim3(256 * 1024 / 1024), 256, 0, stream>>>(w_ds1, ds1h, ds1l, 256 * 1024);
  split_bf16_kernel<<<dim3(256 * 6400 / 1024), 256, 0, stream>>>(w_emb, embh, embl, 256 * 6400);

  // 1) img = image_feature * w_ds2^T + b_ds2   (M=230400, K=768) -> [b][s][c]
  gemm_wmma_bf16x3<1, 0><<<dim3(BT * SS / 32), 256, 0, stream>>>(image, ds2h, ds2l, b_ds2, img, BT * SS, 768);
  // 2) roi_ds = roi_feature * w_ds1^T + b_ds1  (M=19200, K=1024) -> [rr][c][p]
  gemm_wmma_bf16x3<2, 2><<<dim3(ROWS * 25 / 32), 256, 0, stream>>>(roi, ds1h, ds1l, b_ds1, roids, ROWS * 25, 1024);
  // 3) emb = roi_ds * w_emb^T + b_emb          (M=768, K=6400)
  gemm_wmma_bf16x3<0, 0><<<dim3(ROWS / 32), 256, 0, stream>>>(roids, embh, embl, b_emb, embw, ROWS, 6400);
  // 4) attention scores, softmax, context
  scores_kernel<<<dim3(8, BT), 384, 0, stream>>>(img, embw, sc);
  softmax_kernel<<<dim3(ROWS), 256, 0, stream>>>(sc);
  ctx_kernel<<<dim3(4, BT), 256, 0, stream>>>(sc, img, ctxw);
  // 5) LN -> FFN -> LN
  ffn_kernel<<<dim3(ROWS), 256, 0, stream>>>(ctxw, embw, g1, be1, W1, b1, W2, b2, g2, be2, out);
}
